// Attention_56959856280302
// MI455X (gfx1250) — matrix operand order repaired and checked by reading, not compiled, not run on hardware
//
#include <hip/hip_runtime.h>
#include <hip/hip_bf16.h>

// ---------------------------------------------------------------------------
// Types for CDNA5 WMMA (wave32): v_wmma_f32_16x16x32_bf16
// ---------------------------------------------------------------------------
typedef __bf16 bf16;
typedef bf16  bf16x16 __attribute__((ext_vector_type(16)));
typedef bf16  bf16x8  __attribute__((ext_vector_type(8)));
typedef float v8f     __attribute__((ext_vector_type(8)));
typedef int   v4i     __attribute__((vector_size(16)));   // matches builtin param

#define AS1 __attribute__((address_space(1)))
#define AS3 __attribute__((address_space(3)))

// Async global->LDS copy (CDNA5 GLOBAL_LOAD_ASYNC_TO_LDS_B128, ASYNCcnt).
// ROCm 7.2 signature: (v4i addrspace(1)* gsrc, v4i addrspace(3)* ldst,
//                      imm int offset, imm int cpol)
#if __has_builtin(__builtin_amdgcn_global_load_async_to_lds_b128)
#define HAVE_ASYNC_LDS 1
#define ASYNC_CP_B128(dst, src, off)                                        \
    __builtin_amdgcn_global_load_async_to_lds_b128(                         \
        (AS1 v4i*)(void*)(src), (AS3 v4i*)(void*)(dst), (off), 0)
#if __has_builtin(__builtin_amdgcn_s_wait_asynccnt)
#define WAIT_ASYNC0() __builtin_amdgcn_s_wait_asynccnt(0)
#else
#define WAIT_ASYNC0() asm volatile("s_wait_asynccnt 0" ::: "memory")
#endif
#else
#define HAVE_ASYNC_LDS 0
#define WAIT_ASYNC0()
#endif

__device__ __forceinline__ v8f wmma_bf16(bf16x16 a, bf16x16 b, v8f c) {
    // (neg_a, A, neg_b, B, c_mod, C, reuse_a, reuse_b)
    return __builtin_amdgcn_wmma_f32_16x16x32_bf16(false, a, false, b, (short)0, c,
                                                   false, false);
}

__device__ __forceinline__ v8f zero8() {
    v8f z;
#pragma unroll
    for (int i = 0; i < 8; ++i) z[i] = 0.f;
    return z;
}

// Problem constants
constexpr int Bb = 4, T = 2048, Nh = 16, Kh = 4, Gg = 4, Hd = 128;
constexpr int M_rows = Bb * T;                 // 8192
constexpr int Cdim   = Nh * Hd;                // 2048
constexpr float LN1E4 = 9.2103403719761836f;   // ln(10000)

// ---------------------------------------------------------------------------
// One-time conversions (bandwidth-trivial: ~µs at 23.3 TB/s)
// ---------------------------------------------------------------------------
__global__ __launch_bounds__(256) void cvt_bf16(const float* __restrict__ in,
                                                bf16* __restrict__ out) {
    const size_t i = ((size_t)blockIdx.x * 256 + threadIdx.x) * 8;
#pragma unroll
    for (int e = 0; e < 8; ++e) out[i + e] = (bf16)in[i + e];
}

// f32 [Kd x N]  ->  bf16 [N x Kd] (transposed), 32x32 LDS tiles
__global__ __launch_bounds__(256) void cvt_transpose(const float* __restrict__ in,
                                                     bf16* __restrict__ out,
                                                     int Kd, int N) {
    __shared__ bf16 tile[32][33];
    const int tx = threadIdx.x & 31, ty = threadIdx.x >> 5;   // 32 x 8
    const int n0 = blockIdx.x * 32, k0 = blockIdx.y * 32;
#pragma unroll
    for (int i = 0; i < 32; i += 8)
        tile[ty + i][tx] = (bf16)in[(size_t)(k0 + ty + i) * N + n0 + tx];
    __syncthreads();
#pragma unroll
    for (int i = 0; i < 32; i += 8)
        out[(size_t)(n0 + ty + i) * Kd + k0 + tx] = tile[tx][ty + i];
}

// ---------------------------------------------------------------------------
// bf16 WMMA GEMM:  C[M,N] (f32) = A[M,Kd] @ BT[N,Kd]^T
// Block tile 64x128, BK=32; 8 waves (2x4), each wave 32x32 (2x2 frags).
// Tiles staged global->LDS via async copy (no VGPR roundtrip, no convert).
// ---------------------------------------------------------------------------
__global__ __launch_bounds__(256) void gemm_bf16(const bf16* __restrict__ A,
                                                 const bf16* __restrict__ BT,
                                                 float* __restrict__ C,
                                                 int M, int Kd, int N) {
    constexpr int BM = 64, BN = 128, BK = 32;
    constexpr int AST = 40, BST = 40;          // padded strides (80B rows)
    __shared__ bf16 As[BM][AST];               // [row][k]
    __shared__ bf16 BsT[BN][BST];              // [n][k]

    const int tid  = threadIdx.x;
    const int lane = tid & 31;
    const int wave = tid >> 5;
    const int wr = wave >> 2;                  // 0..1
    const int wc = wave & 3;                   // 0..3
    const int tileM = blockIdx.x * BM;
    const int tileN = blockIdx.y * BN;

    const int lh     = lane & 15;
    const int hiHalf = lane >> 4;
    const int kbA    = hiHalf * 8;             // A frag K base {0,8}
    const int kr0    = hiHalf * 16;            // unused; B frags also index k via kbA

    v8f acc[2][2];
#pragma unroll
    for (int i = 0; i < 2; ++i)
#pragma unroll
        for (int j = 0; j < 2; ++j) acc[i][j] = zero8();

    const int aRow = tid >> 2;                 // 0..63
    const int aCol = (tid & 3) * 8;            // 0,8,16,24
    const int bRow = tid >> 1;                 // 0..127
    const int bCol = (tid & 1) * 16;           // 0 | 16

    for (int k0 = 0; k0 < Kd; k0 += BK) {
        const bf16* asrc = A  + (size_t)(tileM + aRow) * Kd + k0 + aCol;
        const bf16* bsrc = BT + (size_t)(tileN + bRow) * Kd + k0 + bCol;
#if HAVE_ASYNC_LDS
        ASYNC_CP_B128(&As[aRow][aCol], asrc, 0);
        ASYNC_CP_B128(&BsT[bRow][bCol], bsrc, 0);
        ASYNC_CP_B128(&BsT[bRow][bCol], bsrc, 16);
#else
        *(bf16x8*)&As[aRow][aCol]      = *(const bf16x8*)asrc;
        *(bf16x8*)&BsT[bRow][bCol]     = *(const bf16x8*)bsrc;
        *(bf16x8*)&BsT[bRow][bCol + 8] = *(const bf16x8*)(bsrc + 8);
#endif
        if (k0 + BK < Kd) {
            __builtin_prefetch(asrc + BK, 0, 0);       // global_prefetch_b8
            __builtin_prefetch(bsrc + BK, 0, 0);
        }
        WAIT_ASYNC0();
        __syncthreads();

        // ---- fragments per ISA VGPR layout, 4 WMMAs ----
        bf16x16 afr[2], bfr[2];
#pragma unroll
        for (int f = 0; f < 2; ++f) {
            const int row = wr * 32 + f * 16 + lh;
            const bf16* p = &As[row][0];
            bf16x8 lo = *(const bf16x8*)(p + kbA);        // K = kbA .. +7
            bf16x8 hi = *(const bf16x8*)(p + kbA + 16);   // K = kbA+16 .. +23
#pragma unroll
            for (int e = 0; e < 8; ++e) { afr[f][e] = lo[e]; afr[f][e + 8] = hi[e]; }
        }
#pragma unroll
        for (int f = 0; f < 2; ++f) {
            const int col = wc * 32 + f * 16 + lh;
            const bf16* p = &BsT[col][0];
            bf16x8 lo = *(const bf16x8*)(p + kbA);
            bf16x8 hi = *(const bf16x8*)(p + kbA + 16);
#pragma unroll
            for (int e = 0; e < 8; ++e) { bfr[f][e] = lo[e]; bfr[f][e + 8] = hi[e]; }
        }
#pragma unroll
        for (int i = 0; i < 2; ++i)
#pragma unroll
            for (int j = 0; j < 2; ++j) acc[i][j] = wmma_bf16(afr[i], bfr[j], acc[i][j]);

        __syncthreads();
    }

    // ---- epilogue: C/D rows in VGPR elems (r | 8+r per lane half) ----
#pragma unroll
    for (int i = 0; i < 2; ++i)
#pragma unroll
        for (int j = 0; j < 2; ++j) {
            const int colBase = tileN + wc * 32 + j * 16 + lh;
            const int rowBase = tileM + wr * 32 + i * 16 + hiHalf * 8;
#pragma unroll
            for (int r = 0; r < 8; ++r)
                C[(size_t)(rowBase + r) * N + colBase] = acc[i][j][r];
        }
}

// ---------------------------------------------------------------------------
// RMSNorm + RoPE + 1/sqrt(H) for q ; bf16 out laid out [b][n][t][h]
// ---------------------------------------------------------------------------
__global__ __launch_bounds__(256) void rms_rope_q(const float* __restrict__ qraw,
                                                  bf16* __restrict__ qb) {
    const int row = blockIdx.x;                 // b*T + t
    const int b = row / T, t = row % T;
    const int tid = threadIdx.x;
    const float* x = qraw + (size_t)row * Cdim;

    float ss = 0.f;
#pragma unroll
    for (int i = 0; i < 8; ++i) { float v = x[tid + i * 256]; ss += v * v; }
    __shared__ float red[256];
    red[tid] = ss;
    __syncthreads();
    for (int s = 128; s > 0; s >>= 1) {
        if (tid < s) red[tid] += red[tid + s];
        __syncthreads();
    }
    const float rinv   = rsqrtf(red[0] / (float)Cdim + 1e-6f);
    const float qscale = rsqrtf((float)Hd);

#pragma unroll
    for (int i = 0; i < 8; ++i) {
        const int c = tid + i * 256;
        const int n = c >> 7, h = c & 127;
        const int hh = h & 63;
        const float inv_ts = __expf(-((float)hh * (1.0f / 64.0f)) * LN1E4);
        float sn, cs;
        __sincosf((float)t * inv_ts, &sn, &cs);
        const float val = x[c] * rinv;
        const int cp = (h < 64) ? c + 64 : c - 64;
        const float rot = ((h < 64) ? -x[cp] : x[cp]) * rinv;
        qb[(((size_t)b * Nh + n) * T + t) * Hd + h] = (bf16)((val * cs + rot * sn) * qscale);
    }
}

// RMSNorm + RoPE for k ; bf16 out [b][k][t][h]
__global__ __launch_bounds__(128) void rms_rope_k(const float* __restrict__ kraw,
                                                  bf16* __restrict__ kb) {
    const int row = blockIdx.x;
    const int b = row / T, t = row % T;
    const int tid = threadIdx.x;
    const float* x = kraw + (size_t)row * (Kh * Hd);

    float ss = 0.f;
#pragma unroll
    for (int i = 0; i < 4; ++i) { float v = x[tid + i * 128]; ss += v * v; }
    __shared__ float red[128];
    red[tid] = ss;
    __syncthreads();
    for (int s = 64; s > 0; s >>= 1) {
        if (tid < s) red[tid] += red[tid + s];
        __syncthreads();
    }
    const float rinv = rsqrtf(red[0] / (float)(Kh * Hd) + 1e-6f);

#pragma unroll
    for (int i = 0; i < 4; ++i) {
        const int c = tid + i * 128;
        const int k = c >> 7, h = c & 127;
        const int hh = h & 63;
        const float inv_ts = __expf(-((float)hh * (1.0f / 64.0f)) * LN1E4);
        float sn, cs;
        __sincosf((float)t * inv_ts, &sn, &cs);
        const float val = x[c] * rinv;
        const int cp = (h < 64) ? c + 64 : c - 64;
        const float rot = ((h < 64) ? -x[cp] : x[cp]) * rinv;
        kb[(((size_t)b * Kh + k) * T + t) * Hd + h] = (bf16)(val * cs + rot * sn);
    }
}

// v: f32 (b,t,k,h) -> bf16 [b][k][t][h]
__global__ __launch_bounds__(256) void cvt_v(const float* __restrict__ vraw,
                                             bf16* __restrict__ vb) {
    const size_t idx = (size_t)blockIdx.x * 256 + threadIdx.x;
    const int h = (int)(idx & 127);
    const size_t r = idx >> 7;
    const int k = (int)(r & 3);
    const size_t bt = r >> 2;
    const int t = (int)(bt % T);
    const int b = (int)(bt / T);
    vb[(((size_t)b * Kh + k) * T + t) * Hd + h] = (bf16)vraw[idx];
}

// ---------------------------------------------------------------------------
// Fused flash attention. Grid (T/64, G, B*K), 128 threads (4 waves).
// Each wave owns 16 query rows; 32-wide s-chunks; online softmax in registers.
// Per chunk per wave: 8 WMMA (q@k^T over H=128) + 8 WMMA (p@v over 128 cols).
// K tile staged via async global->LDS; V staged with in-register transpose.
// ---------------------------------------------------------------------------
__global__ __launch_bounds__(128) void attn_fused(const bf16* __restrict__ qb,
                                                  const bf16* __restrict__ kb,
                                                  const bf16* __restrict__ vb,
                                                  bf16* __restrict__ enc) {
    constexpr int SC = 32;          // s-chunk
    constexpr int KST = 136;        // Ks row stride (h), 272B = 17*16B
    constexpr int VST = 40;         // Vs row stride (s), 80B
    constexpr int PST = 40;
    __shared__ bf16 Ks[SC][KST];    // [s][h]
    __shared__ bf16 Vs[Hd][VST];    // [h][s]  (transposed)
    __shared__ bf16 Pw[4][16][PST]; // per-wave probs transpose scratch

    const int tid  = threadIdx.x;
    const int lane = tid & 31;
    const int wave = tid >> 5;
    const int lh = lane & 15;
    const int hiHalf = lane >> 4;
    const int kbA = hiHalf * 8;
    const int kr0 = hiHalf * 16;

    const int qt0 = blockIdx.x * 64 + wave * 16;
    const int g  = blockIdx.y;
    const int bk = blockIdx.z;
    const int b = bk / Kh, k = bk % Kh;
    const int n = k * Gg + g;

    // preload q fragments (16 rows x 128 H -> 4 A-frags)
    const bf16* qbase = qb + ((size_t)b * Nh + n) * T * Hd;
    bf16x16 qf[4];
#pragma unroll
    for (int hc = 0; hc < 4; ++hc) {
        const bf16* p = qbase + (size_t)(qt0 + lh) * Hd + hc * 32 + kbA;
        bf16x8 lo = *(const bf16x8*)(p);
        bf16x8 hi = *(const bf16x8*)(p + 16);
#pragma unroll
        for (int e = 0; e < 8; ++e) { qf[hc][e] = lo[e]; qf[hc][e + 8] = hi[e]; }
    }

    v8f encAcc[8];
#pragma unroll
    for (int i = 0; i < 8; ++i) encAcc[i] = zero8();
    v8f mrow, lrow;
#pragma unroll
    for (int r = 0; r < 8; ++r) { mrow[r] = -1e30f; lrow[r] = 0.f; }

    const bf16* kbase = kb + ((size_t)b * Kh + k) * T * Hd;
    const bf16* vbase = vb + ((size_t)b * Kh + k) * T * Hd;

    const int srow = tid >> 2;        // 0..31
    const int hb = (tid & 3) * 32;    // 0,32,64,96

    for (int s0 = 0; s0 < T; s0 += SC) {
        // ---- stage K (s-major, async) and V (h-major transpose) ----
        {
            const bf16* ksrc = kbase + (size_t)(s0 + srow) * Hd + hb;
            const bf16* vsrc = vbase + (size_t)(s0 + srow) * Hd + hb;
#if HAVE_ASYNC_LDS
            ASYNC_CP_B128(&Ks[srow][hb], ksrc, 0);
            ASYNC_CP_B128(&Ks[srow][hb], ksrc, 16);
            ASYNC_CP_B128(&Ks[srow][hb], ksrc, 32);
            ASYNC_CP_B128(&Ks[srow][hb], ksrc, 48);
#else
#pragma unroll
            for (int c = 0; c < 4; ++c)
                *(bf16x8*)&Ks[srow][hb + c * 8] = *(const bf16x8*)(ksrc + c * 8);
#endif
            bf16x8 vv[4];
#pragma unroll
            for (int c = 0; c < 4; ++c) vv[c] = *(const bf16x8*)(vsrc + c * 8);
#pragma unroll
            for (int e = 0; e < 32; ++e) Vs[hb + e][srow] = vv[e >> 3][e & 7];
        }
        WAIT_ASYNC0();
        __syncthreads();

        // ---- logits = q @ k^T  (two 16x16 tiles over 32 s-cols) ----
        v8f accL[2];
        accL[0] = zero8();
        accL[1] = zero8();
#pragma unroll
        for (int st = 0; st < 2; ++st) {
#pragma unroll
            for (int hc = 0; hc < 4; ++hc) {
                bf16x16 bf;
                const bf16* p = &Ks[st * 16 + lh][hc * 32 + kbA];
                bf16x8 lo = *(const bf16x8*)(p);
                bf16x8 hi = *(const bf16x8*)(p + 16);
#pragma unroll
                for (int e = 0; e < 8; ++e) { bf[e] = lo[e]; bf[e + 8] = hi[e]; }
                accL[st] = wmma_bf16(qf[hc], bf, accL[st]);
            }
        }

        // ---- online softmax (rows in VGPR elems, cols across 16-lane half) --
        v8f mnew, alpha;
#pragma unroll
        for (int r = 0; r < 8; ++r) {
            float mx = fmaxf(accL[0][r], accL[1][r]);
            mx = fmaxf(mx, __shfl_xor(mx, 1, 16));
            mx = fmaxf(mx, __shfl_xor(mx, 2, 16));
            mx = fmaxf(mx, __shfl_xor(mx, 4, 16));
            mx = fmaxf(mx, __shfl_xor(mx, 8, 16));
            mnew[r]  = fmaxf(mrow[r], mx);
            alpha[r] = __expf(mrow[r] - mnew[r]);
        }
#pragma unroll
        for (int r = 0; r < 8; ++r) {
            accL[0][r] = __expf(accL[0][r] - mnew[r]);
            accL[1][r] = __expf(accL[1][r] - mnew[r]);
            float sm = accL[0][r] + accL[1][r];
            sm += __shfl_xor(sm, 1, 16);
            sm += __shfl_xor(sm, 2, 16);
            sm += __shfl_xor(sm, 4, 16);
            sm += __shfl_xor(sm, 8, 16);
            lrow[r] = lrow[r] * alpha[r] + sm;
            mrow[r] = mnew[r];
        }
#pragma unroll
        for (int o = 0; o < 8; ++o)
#pragma unroll
            for (int r = 0; r < 8; ++r) encAcc[o][r] *= alpha[r];

        // ---- probs (C layout) -> per-wave LDS -> A fragment (transpose) ----
#pragma unroll
        for (int st = 0; st < 2; ++st)
#pragma unroll
            for (int r = 0; r < 8; ++r)
                Pw[wave][hiHalf * 8 + r][st * 16 + lh] = (bf16)accL[st][r];
        asm volatile("s_wait_dscnt 0" ::: "memory");   // same-wave LDS RAW
        bf16x16 pf;
        {
            const bf16* p = &Pw[wave][lh][kbA];
            bf16x8 lo = *(const bf16x8*)(p);
            bf16x8 hi = *(const bf16x8*)(p + 16);
#pragma unroll
            for (int e = 0; e < 8; ++e) { pf[e] = lo[e]; pf[e + 8] = hi[e]; }
        }

        // ---- enc += P(16x32) @ V(32x128), one K=32 WMMA per 16-col chunk ----
#pragma unroll
        for (int oc = 0; oc < 8; ++oc) {
            bf16x16 bf;
            const bf16* p = &Vs[oc * 16 + lh][kbA];
            bf16x8 lo = *(const bf16x8*)(p);
            bf16x8 hi = *(const bf16x8*)(p + 16);
#pragma unroll
            for (int e = 0; e < 8; ++e) { bf[e] = lo[e]; bf[e + 8] = hi[e]; }
            encAcc[oc] = wmma_bf16(pf, bf, encAcc[oc]);
        }
        __syncthreads();
    }

    // ---- normalize and store enc as bf16 at (b*T+t, n*128+h) ----
    v8f inv;
#pragma unroll
    for (int r = 0; r < 8; ++r) inv[r] = 1.0f / lrow[r];
    bf16* obase = enc + ((size_t)b * T) * Cdim + (size_t)n * Hd;
#pragma unroll
    for (int oc = 0; oc < 8; ++oc) {
        const int col = oc * 16 + lh;
#pragma unroll
        for (int r = 0; r < 8; ++r) {
            const int trow = qt0 + hiHalf * 8 + r;
            obase[(size_t)trow * Cdim + col] = (bf16)(encAcc[oc][r] * inv[r]);
        }
    }
}

// ---------------------------------------------------------------------------
// Host launcher
// ---------------------------------------------------------------------------
extern "C" void kernel_launch(void* const* d_in, const int* in_sizes, int n_in,
                              void* d_out, int out_size, void* d_ws, size_t ws_size,
                              hipStream_t stream) {
    const float* x    = (const float*)d_in[0];
    const float* Wq   = (const float*)d_in[1];
    const float* Wk   = (const float*)d_in[2];
    const float* Wv   = (const float*)d_in[3];
    const float* Wout = (const float*)d_in[4];
    float* out = (float*)d_out;
    (void)in_sizes; (void)n_in; (void)out_size; (void)ws_size;

    char* ws = (char*)d_ws;
    size_t off = 0;
    auto carve = [&](size_t bytes) -> void* {
        void* p = ws + off;
        off += (bytes + 255) & ~(size_t)255;
        return p;
    };
    bf16*  xb   = (bf16*)carve((size_t)M_rows * 2048 * 2);
    bf16*  WqT  = (bf16*)carve((size_t)2048 * 2048 * 2);
    bf16*  WkT  = (bf16*)carve((size_t)512 * 2048 * 2);
    bf16*  WvT  = (bf16*)carve((size_t)512 * 2048 * 2);
    bf16*  WoT  = (bf16*)carve((size_t)2048 * 2048 * 2);
    float* qraw = (float*)carve((size_t)M_rows * 2048 * 4);
    float* kraw = (float*)carve((size_t)M_rows * 512 * 4);
    float* vraw = (float*)carve((size_t)M_rows * 512 * 4);
    bf16*  qb   = (bf16*)carve((size_t)M_rows * 2048 * 2);
    bf16*  kb   = (bf16*)carve((size_t)M_rows * 512 * 2);
    bf16*  vb   = (bf16*)carve((size_t)M_rows * 512 * 2);
    bf16*  encb = (bf16*)qraw;   // reuse: qraw dead after rms_rope_q

    // one-time bf16 conversions (+ weight transposes to N-major)
    cvt_bf16<<<(M_rows * 2048) / (256 * 8), 256, 0, stream>>>(x, xb);
    cvt_transpose<<<dim3(64, 64), 256, 0, stream>>>(Wq, WqT, 2048, 2048);
    cvt_transpose<<<dim3(16, 64), 256, 0, stream>>>(Wk, WkT, 2048, 512);
    cvt_transpose<<<dim3(16, 64), 256, 0, stream>>>(Wv, WvT, 2048, 512);
    cvt_transpose<<<dim3(64, 64), 256, 0, stream>>>(Wout, WoT, 2048, 2048);

    dim3 blk(256);
    gemm_bf16<<<dim3(M_rows / 64, 2048 / 128), blk, 0, stream>>>(xb, WqT, qraw, M_rows, 2048, 2048);
    gemm_bf16<<<dim3(M_rows / 64, 512 / 128),  blk, 0, stream>>>(xb, WkT, kraw, M_rows, 2048, 512);
    gemm_bf16<<<dim3(M_rows / 64, 512 / 128),  blk, 0, stream>>>(xb, WvT, vraw, M_rows, 2048, 512);

    rms_rope_q<<<M_rows, 256, 0, stream>>>(qraw, qb);
    rms_rope_k<<<M_rows, 128, 0, stream>>>(kraw, kb);
    cvt_v<<<(M_rows * 512) / 256, 256, 0, stream>>>(vraw, vb);

    attn_fused<<<dim3(T / 64, Gg, Bb * Kh), 128, 0, stream>>>(qb, kb, vb, encb);

    gemm_bf16<<<dim3(M_rows / 64, 2048 / 128), blk, 0, stream>>>(encb, WoT, out, M_rows, 2048, 2048);
}
